// Model_884763263750
// MI455X (gfx1250) — matrix operand order repaired and checked by reading, not compiled, not run on hardware
//
#include <hip/hip_runtime.h>
#include <hip/hip_bf16.h>
#include <math.h>

typedef __attribute__((ext_vector_type(16))) _Float16 v16h;
typedef __attribute__((ext_vector_type(8)))  _Float16 h8;
typedef __attribute__((ext_vector_type(8)))  float    v8f;
typedef __attribute__((ext_vector_type(4)))  int      i4;

#define BN_INV 0.99999500003749969f   // 1/sqrt(1+1e-5)
#define CONV_MAXK 384                 // max padded K (Cin*Kh*Kw) for LDS staging

// branchless selu: exp argument clamped so both sides are always safe;
// lowers to v_exp + v_cndmask, no exec-mask branching.
__device__ __forceinline__ float seluf(float x) {
  const float s = 1.0507009873554805f;
  const float sa = 1.7580993408473766f;          // s * alpha
  float e = __expf(fminf(x, 0.f));
  float neg = sa * (e - 1.f);
  float pos = s * x;
  return x > 0.f ? pos : neg;
}
__device__ __forceinline__ float geluf(float x) {   // exact gelu
  return 0.5f * x * (1.f + erff(x * 0.7071067811865475f));
}
__device__ __forceinline__ float sigm(float x) { return 1.f / (1.f + __expf(-x)); }

// ---------------------------------------------------------------------------
// Direct NCHW conv (only for tiny-K Cin==1 convs: w1 of block0 and its wd).
// ---------------------------------------------------------------------------
__global__ void conv2d_direct_kernel(
    const float* __restrict__ x, const float* __restrict__ w,
    const float* __restrict__ bias, float* __restrict__ y,
    int B, int Cin, int H, int W, int Cout, int Kh, int Kw,
    int padH, int padW, int Hout, int Wout,
    int in_selu, int out_bn, const float* __restrict__ bns, const float* __restrict__ bnb)
{
  int idx = blockIdx.x * blockDim.x + threadIdx.x;
  int total = B * Cout * Hout * Wout;
  if (idx >= total) return;
  int ow = idx % Wout; int t = idx / Wout;
  int oh = t % Hout; t /= Hout;
  int co = t % Cout; int b = t / Cout;
  float acc = bias ? bias[co] : 0.f;
  for (int ci = 0; ci < Cin; ++ci)
    for (int kh = 0; kh < Kh; ++kh) {
      int ih = oh - padH + kh;
      if (ih < 0 || ih >= H) continue;
      for (int kw = 0; kw < Kw; ++kw) {
        int iw = ow - padW + kw;
        if (iw < 0 || iw >= W) continue;
        float v = x[(((size_t)b * Cin + ci) * H + ih) * W + iw];
        if (in_selu) v = seluf(v);
        acc += v * w[((co * Cin + ci) * Kh + kh) * Kw + kw];
      }
    }
  if (out_bn) acc = seluf(acc * (bns[co] * BN_INV) + bnb[co]);
  y[idx] = acc;
}

// ---------------------------------------------------------------------------
// Implicit-GEMM conv via V_WMMA_F32_16X16X32_F16.
// 256 threads = 8 waves; block owns one 16-channel tile and 8 16-pixel tiles.
// LDS staged once per block: f16 weights (zero-padded to KP), flat gather
// offsets, and packed (kh,kw) for edge checks. Interior/edge branch is
// scalarized via readfirstlane; both gather paths are two-phase
// (issue all loads, then activate+convert) so loads batch under staggered
// waits. K loop unrolled x2 so two WMMA chunks' loads are in flight.
// ---------------------------------------------------------------------------
__global__ void conv2d_wmma_kernel(
    const float* __restrict__ x, const float* __restrict__ w,
    const float* __restrict__ bias, float* __restrict__ y,
    int B, int Cin, int H, int W, int Cout, int Kh, int Kw,
    int padH, int padW, int Hout, int Wout,
    int in_selu, int out_bn, const float* __restrict__ bns, const float* __restrict__ bnb,
    int tilesM)
{
  __shared__ __align__(16) _Float16 As[16 * CONV_MAXK];
  __shared__ __align__(16) int Offs[CONV_MAXK];
  __shared__ __align__(16) int Khw[CONV_MAXK];

  const int tid  = threadIdx.x;
  const int lane = tid & 31;
  const int wid  = tid >> 5;            // wave id in block: 0..7
  const int half = (lane >> 4) & 1;
  const int l16  = lane & 15;
  const int tm   = blockIdx.x % tilesM;
  const int tnb  = blockIdx.x / tilesM;

  const int KK = Cin * Kh * Kw;
  const int KP = (KK + 31) & ~31;       // padded K (<= CONV_MAXK)
  const int HW = H * W;
  const int KhKw = Kh * Kw;

  // ---- stage k -> offset tables (division done once, cooperatively)
  for (int k = tid; k < KP; k += 256) {
    int off = 0, packed = 0;
    if (k < KK) {
      int ci = k / KhKw;
      int rm = k - ci * KhKw;
      int kh = rm / Kw;
      int kw = rm - kh * Kw;
      off = ci * HW + kh * W + kw;
      packed = (kh << 16) | kw;
    }
    Offs[k] = off;
    Khw[k]  = packed;
  }
  // ---- stage weight tile (16 x KP, zero-padded) into LDS as f16
  for (int idx = tid; idx < 16 * KP; idx += 256) {
    int row = idx / KP, col = idx - row * KP;
    int gr = tm * 16 + row;
    float v = (gr < Cout && col < KK) ? w[(size_t)gr * KK + col] : 0.f;
    As[row * KP + col] = (_Float16)v;
  }
  __syncthreads();

  // ---- this wave's pixel tile
  const int Npix = B * Hout * Wout;
  const int tn = tnb * 8 + wid;
  const int n  = tn * 16 + l16;
  const bool nvalid = n < Npix;
  const int nc = nvalid ? n : (Npix - 1);
  const int pb   = nc / (Hout * Wout);
  const int rrem = nc - pb * (Hout * Wout);
  const int poh  = rrem / Wout;
  const int pow_ = rrem - poh * Wout;

  // wave-uniform interior test; scalarize with readfirstlane so the
  // interior/edge branch is an s_cbranch, not an exec-mask dance.
  // Interior fast path is safe on the K tail chunk: As is zero beyond K and
  // Offs is clamped to 0 there, so those products vanish.
  const int t0 = tn * 16;
  int interior_i = 0;
  if (t0 + 15 < Npix) {
    int pb0 = t0 / (Hout * Wout);
    int rr  = t0 - pb0 * (Hout * Wout);
    int poh0 = rr / Wout;
    int ow0  = rr - poh0 * Wout;
    interior_i = (ow0 + 15 < Wout)
              && (poh0 >= padH) && (poh0 - padH + Kh <= H)
              && (ow0 >= padW) && (ow0 + 15 - padW + Kw <= W);
  }
  const bool interior = __builtin_amdgcn_readfirstlane(interior_i) != 0;

  const _Float16* arow = As + l16 * KP;
  // absolute flat base index of this pixel's patch origin (may be "negative"
  // for edge pixels; only used after validity clamp in the edge path)
  const int babs = pb * Cin * HW + (poh - padH) * W + (pow_ - padW);
  const float* bbase = x + (ptrdiff_t)babs;

  v8f acc = {0.f, 0.f, 0.f, 0.f, 0.f, 0.f, 0.f, 0.f};

  auto do_chunk = [&](int k0) {
    // ---- A fragment: two aligned 16B LDS loads
    const int ra = k0 + 16 * half;
    h8 alo = *(const h8*)(arow + ra);
    h8 ahi = *(const h8*)(arow + ra + 8);
    v16h av;
#pragma unroll
    for (int j = 0; j < 8; ++j) { av[j] = alo[j]; av[j + 8] = ahi[j]; }

    // ---- gather offsets for this chunk (contiguous, 16B-aligned)
    const int kks = k0 + 16 * half;
    i4 o0 = *(const i4*)(Offs + kks);
    i4 o1 = *(const i4*)(Offs + kks + 4);
    i4 o2 = *(const i4*)(Offs + kks + 8);
    i4 o3 = *(const i4*)(Offs + kks + 12);
    int of[16];
#pragma unroll
    for (int j = 0; j < 4; ++j) {
      of[j] = o0[j]; of[4 + j] = o1[j]; of[8 + j] = o2[j]; of[12 + j] = o3[j];
    }

    float vals[16];
    if (interior) {
      // phase 1: issue all 16 loads (no dependent ALU in between)
#pragma unroll
      for (int j = 0; j < 16; ++j) vals[j] = bbase[of[j]];
    } else {
      // branchless edge path: clamp address to 0 when invalid, select 0.
      i4 q0 = *(const i4*)(Khw + kks);
      i4 q1 = *(const i4*)(Khw + kks + 4);
      i4 q2 = *(const i4*)(Khw + kks + 8);
      i4 q3 = *(const i4*)(Khw + kks + 12);
      int qh[16];
#pragma unroll
      for (int j = 0; j < 4; ++j) {
        qh[j] = q0[j]; qh[4 + j] = q1[j]; qh[8 + j] = q2[j]; qh[12 + j] = q3[j];
      }
      bool okf[16];
#pragma unroll
      for (int j = 0; j < 16; ++j) {
        int kk = kks + j;
        int ih = poh - padH + (qh[j] >> 16);
        int iw = pow_ - padW + (qh[j] & 0xffff);
        bool ok = nvalid & (kk < KK) & (ih >= 0) & (ih < H) & (iw >= 0) & (iw < W);
        okf[j] = ok;
        int idxc = ok ? (babs + of[j]) : 0;     // always a legal address
        vals[j] = x[idxc];
      }
#pragma unroll
      for (int j = 0; j < 16; ++j) vals[j] = okf[j] ? vals[j] : 0.f;
    }
    // phase 2: activation + f16 conversion (selu(0)==0)
    v16h bv;
#pragma unroll
    for (int j = 0; j < 16; ++j) {
      float v = vals[j];
      if (in_selu) v = seluf(v);
      bv[j] = (_Float16)v;
    }
    acc = __builtin_amdgcn_wmma_f32_16x16x32_f16(
        false, av, false, bv, (short)0, acc, false, false);
  };

  // K loop unrolled x2: two chunks of loads in flight per iteration
  int k0 = 0;
  for (; k0 + 64 <= KP; k0 += 64) {
    do_chunk(k0);
    do_chunk(k0 + 32);
  }
  if (k0 < KP) do_chunk(k0);

  if (!nvalid) return;
#pragma unroll
  for (int r = 0; r < 8; ++r) {
    int m = tm * 16 + r + 8 * half;             // D row = output channel
    if (m >= Cout) continue;
    float v = acc[r] + (bias ? bias[m] : 0.f);
    if (out_bn) v = seluf(v * (bns[m] * BN_INV) + bnb[m]);
    y[(((size_t)pb * Cout + m) * Hout + poh) * Wout + pow_] = v;
  }
}

// ---------------------------------------------------------------------------
// Dense layer Y(M,N) = X(M,K) @ Wt(K,N) + b via WMMA f16.  Requires K%32==0.
// Branchless: out-of-range rows/cols are clamped (their D entries are never
// written), so all loads are unconditional; A uses vectorized float4 loads.
// ---------------------------------------------------------------------------
__global__ void linear_wmma_kernel(
    const float* __restrict__ X, const float* __restrict__ Wt,
    const float* __restrict__ bias, float* __restrict__ Y,
    int M, int K, int N)
{
  const int lane = threadIdx.x;
  const int half = lane >> 4;
  const int l16  = lane & 15;
  const int tilesR = (M + 15) >> 4;
  const int tr = blockIdx.x % tilesR;
  const int tc = blockIdx.x / tilesR;
  const int ar = tr * 16 + l16;
  const int bn = tc * 16 + l16;
  const int arc = (ar < M) ? ar : (M - 1);
  const int bnc = (bn < N) ? bn : (N - 1);
  const float* xr = X + (size_t)arc * K;

  v8f acc = {0.f, 0.f, 0.f, 0.f, 0.f, 0.f, 0.f, 0.f};
  for (int k0 = 0; k0 < K; k0 += 32) {
    const int ra = k0 + 16 * half;      // contiguous 16-run
    float4 a0 = *(const float4*)(xr + ra);
    float4 a1 = *(const float4*)(xr + ra + 4);
    float4 a2 = *(const float4*)(xr + ra + 8);
    float4 a3 = *(const float4*)(xr + ra + 12);
    v16h av;
    av[0]=(_Float16)a0.x;  av[1]=(_Float16)a0.y;  av[2]=(_Float16)a0.z;  av[3]=(_Float16)a0.w;
    av[4]=(_Float16)a1.x;  av[5]=(_Float16)a1.y;  av[6]=(_Float16)a1.z;  av[7]=(_Float16)a1.w;
    av[8]=(_Float16)a2.x;  av[9]=(_Float16)a2.y;  av[10]=(_Float16)a2.z; av[11]=(_Float16)a2.w;
    av[12]=(_Float16)a3.x; av[13]=(_Float16)a3.y; av[14]=(_Float16)a3.z; av[15]=(_Float16)a3.w;
    const int kks = k0 + 16 * half;
    float bvals[16];
#pragma unroll
    for (int j = 0; j < 16; ++j)
      bvals[j] = Wt[(size_t)(kks + j) * N + bnc];
    v16h bv;
#pragma unroll
    for (int j = 0; j < 16; ++j)
      bv[j] = (_Float16)bvals[j];
    acc = __builtin_amdgcn_wmma_f32_16x16x32_f16(
        false, av, false, bv, (short)0, acc, false, false);
  }
  if (bn >= N) return;
#pragma unroll
  for (int r = 0; r < 8; ++r) {
    int m = tr * 16 + r + 8 * half;
    if (m >= M) continue;
    Y[(size_t)m * N + bn] = acc[r] + (bias ? bias[bn] : 0.f);
  }
}

// ---------------------------------------------------------------------------
// |x| -> maxpool(3,3) over (filter,time) -> bn1 -> selu
// ---------------------------------------------------------------------------
__global__ void pool33_bn_selu_kernel(
    const float* __restrict__ in, float* __restrict__ out,
    const float* __restrict__ s1, const float* __restrict__ b1,
    int B, int Hin, int Win, int Hp, int Wp)
{
  int idx = blockIdx.x * blockDim.x + threadIdx.x;
  int total = B * Hp * Wp;
  if (idx >= total) return;
  int wp = idx % Wp; int t = idx / Wp;
  int hp = t % Hp; int b = t / Hp;
  float m = -3.0e38f;
  for (int i = 0; i < 3; ++i)
    for (int j = 0; j < 3; ++j) {
      float v = fabsf(in[((size_t)b * Hin + (3 * hp + i)) * Win + (3 * wp + j)]);
      m = fmaxf(m, v);
    }
  out[idx] = seluf(m * (s1[0] * BN_INV) + b1[0]);
}

// y = maxpool_1x3(a + ident), widths Win -> Wout = Win/3
__global__ void add_pool13_kernel(
    const float* __restrict__ a, const float* __restrict__ ident,
    float* __restrict__ y, int total, int Wout, int Win)
{
  int idx = blockIdx.x * blockDim.x + threadIdx.x;
  if (idx >= total) return;
  int wo = idx % Wout;
  size_t row = idx / Wout;
  size_t off = row * Win + 3 * wo;
  float m = -3.0e38f;
  for (int j = 0; j < 3; ++j) m = fmaxf(m, a[off + j] + ident[off + j]);
  y[idx] = m;
}

// e (B,C,Hh,Wd) -> eS (B,Hh,C) = max_w |e| + pos[h,c]
__global__ void reduce_eS_kernel(const float* __restrict__ e, const float* __restrict__ pos,
                                 float* __restrict__ out, int B, int C, int Hh, int Wd)
{
  int idx = blockIdx.x * blockDim.x + threadIdx.x;
  if (idx >= B * Hh * C) return;
  int c = idx % C; int t = idx / C;
  int h = t % Hh; int b = t / Hh;
  const float* p = e + (((size_t)b * C + c) * Hh + h) * Wd;
  float m = 0.f;
  for (int w = 0; w < Wd; ++w) m = fmaxf(m, fabsf(p[w]));
  out[idx] = m + pos[h * C + c];
}

// e (B,C,Hh,Wd) -> eT (B,Wd,C) = max_h |e|
__global__ void reduce_eT_kernel(const float* __restrict__ e, float* __restrict__ out,
                                 int B, int C, int Hh, int Wd)
{
  int idx = blockIdx.x * blockDim.x + threadIdx.x;
  if (idx >= B * Wd * C) return;
  int c = idx % C; int t = idx / C;
  int w = t % Wd; int b = t / Wd;
  float m = 0.f;
  for (int h = 0; h < Hh; ++h)
    m = fmaxf(m, fabsf(e[(((size_t)b * C + c) * Hh + h) * Wd + w]));
  out[idx] = m;
}

// FAN tail: out = [gate*cos(ph), gate*sin(ph), (1-gate)*gelu(gg)]
__global__ void fan_post_kernel(const float* __restrict__ ph, const float* __restrict__ gg,
                                const float* __restrict__ gatep, float* __restrict__ out,
                                int R, int P, int G)
{
  int D = 2 * P + G;
  int idx = blockIdx.x * blockDim.x + threadIdx.x;
  if (idx >= R * D) return;
  int c = idx % D; int r = idx / D;
  float gate = sigm(gatep[0]);
  float v;
  if (c < P)           v = gate * __cosf(ph[(size_t)r * P + c]);
  else if (c < 2 * P)  v = gate * __sinf(ph[(size_t)r * P + (c - P)]);
  else                 v = (1.f - gate) * geluf(gg[(size_t)r * G + (c - 2 * P)]);
  out[idx] = v;
}

// gated top-K pool, descending scores, ties -> lower index (lax.top_k).
__global__ void gpool_kernel(const float* __restrict__ h, const float* __restrict__ w,
                             const float* __restrict__ bias, float* __restrict__ out,
                             int N, int D, int K)
{
  __shared__ float sc[160];
  __shared__ float sc2[160];
  __shared__ int   sidx[128];
  int b = blockIdx.x;
  const float* hb = h + (size_t)b * N * D;
  for (int i = threadIdx.x; i < N; i += blockDim.x) {
    float s = bias[0];
    for (int d = 0; d < D; ++d) s += hb[(size_t)i * D + d] * w[d];
    s = sigm(s);
    sc[i] = s; sc2[i] = s;
  }
  __syncthreads();
  if (threadIdx.x == 0) {
    for (int k = 0; k < K; ++k) {
      float best = -3.f; int bi = 0;
      for (int i = 0; i < N; ++i) if (sc2[i] > best) { best = sc2[i]; bi = i; }
      sidx[k] = bi; sc2[bi] = -4.f;
    }
  }
  __syncthreads();
  for (int t = threadIdx.x; t < K * D; t += blockDim.x) {
    int k = t / D, d = t % D;
    out[((size_t)b * K + k) * D + d] = hb[(size_t)sidx[k] * D + d] * sc[sidx[k]];
  }
}

__global__ void bcast_rows_kernel(const float* __restrict__ src, float* __restrict__ dst,
                                  int R, int D)
{
  int idx = blockIdx.x * blockDim.x + threadIdx.x;
  if (idx < R * D) dst[idx] = src[idx % D];
}

__global__ void add_kernel(const float* __restrict__ a, const float* __restrict__ b,
                           float* __restrict__ c, int total)
{
  int idx = blockIdx.x * blockDim.x + threadIdx.x;
  if (idx < total) c[idx] = a[idx] + b[idx];
}

__global__ void max_kernel(const float* __restrict__ a, const float* __restrict__ b,
                           float* __restrict__ c, int total)
{
  int idx = blockIdx.x * blockDim.x + threadIdx.x;
  if (idx < total) c[idx] = fmaxf(a[idx], b[idx]);
}

// feats(B,5D) = [max|oT|, mean oT, max|oS|, mean oS, m]
__global__ void feats_kernel(const float* __restrict__ oT, const float* __restrict__ oS,
                             const float* __restrict__ m, float* __restrict__ feats,
                             int B, int NT, int NS, int D)
{
  int idx = blockIdx.x * blockDim.x + threadIdx.x;
  if (idx >= B * 5 * D) return;
  int j = idx % (5 * D); int b = idx / (5 * D);
  float v = 0.f;
  if (j < D) {
    for (int i = 0; i < NT; ++i) v = fmaxf(v, fabsf(oT[((size_t)b * NT + i) * D + j]));
  } else if (j < 2 * D) {
    int c = j - D; float s = 0.f;
    for (int i = 0; i < NT; ++i) s += oT[((size_t)b * NT + i) * D + c];
    v = s / (float)NT;
  } else if (j < 3 * D) {
    int c = j - 2 * D;
    for (int i = 0; i < NS; ++i) v = fmaxf(v, fabsf(oS[((size_t)b * NS + i) * D + c]));
  } else if (j < 4 * D) {
    int c = j - 3 * D; float s = 0.f;
    for (int i = 0; i < NS; ++i) s += oS[((size_t)b * NS + i) * D + c];
    v = s / (float)NS;
  } else {
    v = m[(size_t)b * D + (j - 4 * D)];
  }
  feats[idx] = v;
}

// ===========================================================================
// Host side
// ===========================================================================
namespace {

struct Bump {
  float* base; size_t off;
  float* get(size_t n) { float* p = base + off; off += (n + 63) & ~(size_t)63; return p; }
};

struct FanP { const float *bg, *bp, *gate, *wg, *wp; };
struct HtrgP { FanP fan1, fan2; const float *mb, *mw; };

inline int cdiv(int a, int b) { return (a + b - 1) / b; }

void conv_direct(hipStream_t st, const float* x, const float* w, const float* bias, float* y,
                 int B, int Cin, int H, int W, int Cout, int Kh, int Kw,
                 int padH, int padW, int Hout, int Wout,
                 int in_selu, int out_bn, const float* bns, const float* bnb) {
  int total = B * Cout * Hout * Wout;
  conv2d_direct_kernel<<<cdiv(total, 256), 256, 0, st>>>(
      x, w, bias, y, B, Cin, H, W, Cout, Kh, Kw, padH, padW, Hout, Wout,
      in_selu, out_bn, bns, bnb);
}

void conv_wmma(hipStream_t st, const float* x, const float* w, const float* bias, float* y,
               int B, int Cin, int H, int W, int Cout, int Kh, int Kw,
               int padH, int padW, int Hout, int Wout,
               int in_selu, int out_bn, const float* bns, const float* bnb) {
  int tilesM  = cdiv(Cout, 16);
  int tilesNB = cdiv(B * Hout * Wout, 16 * 8);   // 8 pixel-tiles per block
  conv2d_wmma_kernel<<<tilesM * tilesNB, 256, 0, st>>>(
      x, w, bias, y, B, Cin, H, W, Cout, Kh, Kw, padH, padW, Hout, Wout,
      in_selu, out_bn, bns, bnb, tilesM);
}

void linear_w(hipStream_t st, const float* X, const float* Wt, const float* b, float* Y,
              int M, int K, int N) {                 // K must be a multiple of 32
  int g = cdiv(M, 16) * cdiv(N, 16);
  linear_wmma_kernel<<<g, 32, 0, st>>>(X, Wt, b, Y, M, K, N);
}

// FAN layer: X (R,din) -> out (R,dout)
void fan_apply(hipStream_t st, Bump& al, const float* X, int R, int din, int dout,
               const FanP& p, float* out) {
  int P = dout / 4, G = dout - dout / 2;
  float* ph = al.get((size_t)R * P);
  float* gg = al.get((size_t)R * G);
  linear_w(st, X, p.wp, p.bp, ph, R, din, P);
  linear_w(st, X, p.wg, p.bg, gg, R, din, G);
  int total = R * dout;
  fan_post_kernel<<<cdiv(total, 256), 256, 0, st>>>(ph, gg, p.gate, out, R, P, G);
}

} // namespace

extern "C" void kernel_launch(void* const* d_in, const int* in_sizes, int n_in,
                              void* d_out, int out_size, void* d_ws, size_t ws_size,
                              hipStream_t stream) {
  (void)in_sizes; (void)out_size; (void)ws_size;
  const int B = 4;

  // ---- parameter leaves (jax pytree order: sorted dict keys, lists in order)
  const float* L[128];
  for (int i = 0; i < n_in && i < 128; ++i) L[i] = (const float*)d_in[i];
  int p = 0;
  // blocks[0]: b1,b2,bd,bn_b,bn_s,w1,w2,wd
  const float *B0_b1=L[p++], *B0_b2=L[p++], *B0_bd=L[p++], *B0_bnb=L[p++], *B0_bns=L[p++],
              *B0_w1=L[p++], *B0_w2=L[p++], *B0_wd=L[p++];
  // blocks[1]: b1,b2,bn_b,bn_s,w1,w2
  const float *B1_b1=L[p++], *B1_b2=L[p++], *B1_bnb=L[p++], *B1_bns=L[p++],
              *B1_w1=L[p++], *B1_w2=L[p++];
  // blocks[2]
  const float *B2_b1=L[p++], *B2_b2=L[p++], *B2_bd=L[p++], *B2_bnb=L[p++], *B2_bns=L[p++],
              *B2_w1=L[p++], *B2_w2=L[p++], *B2_wd=L[p++];
  // blocks[3]
  const float *B3_b1=L[p++], *B3_b2=L[p++], *B3_bnb=L[p++], *B3_bns=L[p++],
              *B3_w1=L[p++], *B3_w2=L[p++];
  const float *bn1_b=L[p++], *bn1_s=L[p++];
  FanP gatS = { L[p], L[p+1], L[p+2], L[p+3], L[p+4] }; p += 5;   // bg,bp,gate,wg,wp
  FanP gatT = { L[p], L[p+1], L[p+2], L[p+3], L[p+4] }; p += 5;
  HtrgP ht[4];
  for (int i = 0; i < 4; ++i) {
    ht[i].fan1 = { L[p], L[p+1], L[p+2], L[p+3], L[p+4] }; p += 5;
    ht[i].fan2 = { L[p], L[p+1], L[p+2], L[p+3], L[p+4] }; p += 5;
    ht[i].mb = L[p++]; ht[i].mw = L[p++];
  }
  const float *master1=L[p++], *master2=L[p++];
  const float *out_b=L[p++], *out_w=L[p++];
  const float *poolS_b=L[p++], *poolS_w=L[p++];
  const float *poolT_b=L[p++], *poolT_w=L[p++];
  const float *ph_b[4], *ph_w[4];
  for (int i = 0; i < 4; ++i) { ph_b[i] = L[p++]; ph_w[i] = L[p++]; }
  const float *pos_S=L[p++], *sinc=L[p++], *x=L[p++];

  // ---- workspace plan (floats)
  Bump ws{ (float*)d_ws, 0 };
  const size_t EACT = 10500000;         // ping/pong activation regions
  float* EA = ws.get(EACT);
  float* EB = ws.get(EACT);
  float* T1 = ws.get(33000000);         // conv scratch (also sinc output, head scratch)
  float* T2 = ws.get(31500000);
  float* T3 = ws.get(31500000);

  // ---- stage 1: sinc filterbank conv1d VALID via WMMA: (4,32000)->(4,70,31872)
  const int Wsinc = 32000 - 129 + 1;    // 31872
  conv_wmma(stream, x, sinc, nullptr, T1,
            B, 1, 1, 32000, 70, 1, 129, 0, 0, 1, Wsinc, 0, 0, nullptr, nullptr);

  // ---- stage 2: |.| -> maxpool(3,3) -> bn1 -> selu : (4,70,31872)->(4,23,10624)
  const int H0 = 23, W0 = 10624;
  {
    int total = B * H0 * W0;
    pool33_bn_selu_kernel<<<cdiv(total, 256), 256, 0, stream>>>(
        T1, EA, bn1_s, bn1_b, B, 70, Wsinc, H0, W0);
  }

  // ---- residual blocks ----------------------------------------------------
  // block 0: 1 -> 32 (first: no input selu), widths 10624 -> 3541
  {
    conv_direct(stream, EA, B0_w1, B0_b1, T1, B, 1, 23, W0, 32, 2, 3, 1, 1, 24, W0,
                0, 1, B0_bns, B0_bnb);
    conv_wmma(stream, T1, B0_w2, B0_b2, T2, B, 32, 24, W0, 32, 2, 3, 0, 1, 23, W0,
              0, 0, nullptr, nullptr);
    conv_direct(stream, EA, B0_wd, B0_bd, T3, B, 1, 23, W0, 32, 1, 3, 0, 1, 23, W0,
                0, 0, nullptr, nullptr);
    int Wp = W0 / 3;                    // 3541
    int total = B * 32 * 23 * Wp;
    add_pool13_kernel<<<cdiv(total, 256), 256, 0, stream>>>(T2, T3, EB, total, Wp, W0);
  }
  // block 1: 32 -> 32, widths 3541 -> 1180
  {
    const int Wi = 3541;
    conv_wmma(stream, EB, B1_w1, B1_b1, T1, B, 32, 23, Wi, 32, 2, 3, 1, 1, 24, Wi,
              1, 1, B1_bns, B1_bnb);
    conv_wmma(stream, T1, B1_w2, B1_b2, T2, B, 32, 24, Wi, 32, 2, 3, 0, 1, 23, Wi,
              0, 0, nullptr, nullptr);
    int Wp = Wi / 3;                    // 1180
    int total = B * 32 * 23 * Wp;
    add_pool13_kernel<<<cdiv(total, 256), 256, 0, stream>>>(T2, EB, EA, total, Wp, Wi);
  }
  // block 2: 32 -> 64, widths 1180 -> 393
  {
    const int Wi = 1180;
    conv_wmma(stream, EA, B2_w1, B2_b1, T1, B, 32, 23, Wi, 64, 2, 3, 1, 1, 24, Wi,
              1, 1, B2_bns, B2_bnb);
    conv_wmma(stream, T1, B2_w2, B2_b2, T2, B, 64, 24, Wi, 64, 2, 3, 0, 1, 23, Wi,
              0, 0, nullptr, nullptr);
    conv_wmma(stream, EA, B2_wd, B2_bd, T3, B, 32, 23, Wi, 64, 1, 3, 0, 1, 23, Wi,
              0, 0, nullptr, nullptr);
    int Wp = Wi / 3;                    // 393
    int total = B * 64 * 23 * Wp;
    add_pool13_kernel<<<cdiv(total, 256), 256, 0, stream>>>(T2, T3, EB, total, Wp, Wi);
  }
  // block 3: 64 -> 64, widths 393 -> 131
  {
    const int Wi = 393;
    conv_wmma(stream, EB, B3_w1, B3_b1, T1, B, 64, 23, Wi, 64, 2, 3, 1, 1, 24, Wi,
              1, 1, B3_bns, B3_bnb);
    conv_wmma(stream, T1, B3_w2, B3_b2, T2, B, 64, 24, Wi, 64, 2, 3, 0, 1, 23, Wi,
              0, 0, nullptr, nullptr);
    int Wp = Wi / 3;                    // 131
    int total = B * 64 * 23 * Wp;
    add_pool13_kernel<<<cdiv(total, 256), 256, 0, stream>>>(T2, EB, EA, total, Wp, Wi);
  }
  // e = EA, shape (4,64,23,131)
  const int C = 64, Hh = 23, Wd = 131;

  // ---- head ---------------------------------------------------------------
  Bump hd{ T1, 0 };                     // T1/T2/T3 free now; e lives in EA

  float* eS = hd.get((size_t)B * Hh * C);   // (4,23,64)
  float* eT = hd.get((size_t)B * Wd * C);   // (4,131,64)
  reduce_eS_kernel<<<cdiv(B * Hh * C, 256), 256, 0, stream>>>(EA, pos_S, eS, B, C, Hh, Wd);
  reduce_eT_kernel<<<cdiv(B * Wd * C, 256), 256, 0, stream>>>(EA, eT, B, C, Hh, Wd);

  float* fS = hd.get((size_t)B * Hh * 64);
  float* fT = hd.get((size_t)B * Wd * 64);
  fan_apply(stream, hd, eS, B * Hh, 64, 64, gatS, fS);
  fan_apply(stream, hd, eT, B * Wd, 64, 64, gatT, fT);

  const int KS = 11, KT = 91;           // int(23*0.5), int(131*0.7)
  float* outS = hd.get((size_t)B * KS * 64);
  float* outT = hd.get((size_t)B * KT * 64);
  gpool_kernel<<<B, 128, 0, stream>>>(fS, poolS_w, poolS_b, outS, Hh, 64, KS);
  gpool_kernel<<<B, 128, 0, stream>>>(fT, poolT_w, poolT_b, outT, Wd, 64, KT);

  const int KS2 = 5, KT2 = 45;          // int(11*0.5), int(91*0.5)
  float* br_t[2]; float* br_s[2]; float* br_m[2];

  for (int br = 0; br < 2; ++br) {
    const HtrgP& h0 = ht[br * 2];
    const HtrgP& h1 = ht[br * 2 + 1];
    const float* master = (br == 0) ? master1 : master2;
    const float* phs_w = ph_w[br * 2],     *phs_b = ph_b[br * 2];
    const float* pht_w = ph_w[br * 2 + 1], *pht_b = ph_b[br * 2 + 1];

    float* tb = hd.get((size_t)B * KT * 32);
    float* sb = hd.get((size_t)B * KS * 32);
    fan_apply(stream, hd, outT, B * KT, 64, 32, h0.fan1, tb);
    fan_apply(stream, hd, outS, B * KS, 64, 32, h0.fan2, sb);

    float* mrows = hd.get((size_t)B * 64);
    bcast_rows_kernel<<<cdiv(B * 64, 256), 256, 0, stream>>>(master, mrows, B, 64);
    float* m1 = hd.get((size_t)B * 32);
    linear_w(stream, mrows, h0.mw, h0.mb, m1, B, 64, 32);

    float* sp = hd.get((size_t)B * KS2 * 32);
    float* tp = hd.get((size_t)B * KT2 * 32);
    gpool_kernel<<<B, 128, 0, stream>>>(sb, phs_w, phs_b, sp, KS, 32, KS2);
    gpool_kernel<<<B, 128, 0, stream>>>(tb, pht_w, pht_b, tp, KT, 32, KT2);

    float* ta = hd.get((size_t)B * KT2 * 32);
    float* sa = hd.get((size_t)B * KS2 * 32);
    fan_apply(stream, hd, tp, B * KT2, 32, 32, h1.fan1, ta);
    fan_apply(stream, hd, sp, B * KS2, 32, 32, h1.fan2, sa);
    float* m2 = hd.get((size_t)B * 32);
    linear_w(stream, m1, h1.mw, h1.mb, m2, B, 32, 32);

    br_t[br] = hd.get((size_t)B * KT2 * 32);
    br_s[br] = hd.get((size_t)B * KS2 * 32);
    br_m[br] = hd.get((size_t)B * 32);
    add_kernel<<<cdiv(B * KT2 * 32, 256), 256, 0, stream>>>(tp, ta, br_t[br], B * KT2 * 32);
    add_kernel<<<cdiv(B * KS2 * 32, 256), 256, 0, stream>>>(sp, sa, br_s[br], B * KS2 * 32);
    add_kernel<<<cdiv(B * 32, 256), 256, 0, stream>>>(m1, m2, br_m[br], B * 32);
  }

  float* oT = hd.get((size_t)B * KT2 * 32);
  float* oS = hd.get((size_t)B * KS2 * 32);
  float* mM = hd.get((size_t)B * 32);
  max_kernel<<<cdiv(B * KT2 * 32, 256), 256, 0, stream>>>(br_t[0], br_t[1], oT, B * KT2 * 32);
  max_kernel<<<cdiv(B * KS2 * 32, 256), 256, 0, stream>>>(br_s[0], br_s[1], oS, B * KS2 * 32);
  max_kernel<<<cdiv(B * 32, 256), 256, 0, stream>>>(br_m[0], br_m[1], mM, B * 32);

  float* feats = hd.get((size_t)B * 160);
  feats_kernel<<<cdiv(B * 160, 256), 256, 0, stream>>>(oT, oS, mM, feats, B, KT2, KS2, 32);

  // final logits (4,160)@(160,2)+b -> d_out (4,2)
  linear_w(stream, feats, out_w, out_b, (float*)d_out, B, 160, 2);
}
